// Decoder1_88184268521800
// MI455X (gfx1250) — matrix operand order repaired and checked by reading, not compiled, not run on hardware
//
#include <hip/hip_runtime.h>
#include <math.h>

typedef __bf16 bf16;
typedef __attribute__((ext_vector_type(16))) __bf16 v16bf;
typedef __attribute__((ext_vector_type(8)))  float  v8f;

#define B_   64
#define TD_  512
#define TO_  256
#define A_   256
#define E_   512
#define H1_  512
#define V_   4096
#define KT1  40   // x1 = [ex(512), ctx(256), h1(512)] -> K=1280 -> 40 k-tiles
#define KT2  24   // x2 = [h1(512), h2(256)]           -> K=768  -> 24 k-tiles
#define KTY  16   // y  = [h2(256), ctx(256)]          -> K=512  -> 16 k-tiles

__device__ __forceinline__ float sigmoidf_(float x) { return 1.f / (1.f + expf(-x)); }

// ---- WMMA fragment packing (ISA 7.12.2 layouts, wave32) -------------------
// B-matrix (K x N), 16-bit: col=lane&15; lanes>=16 carry K+8; VGPRs 0-3 -> K 0..7,
// VGPRs 4-7 -> K 16..23 (same order as A). Fragment = 32 lanes x 32B, index (nt*KT+kt).
__device__ __forceinline__ void packB(bf16* base, int KT, int k, int n, float v) {
  int nt = n >> 4, c = n & 15;
  int kt = k >> 5, kk = k & 31;
  int lane = c + ((kk & 8) << 1);
  int j = ((kk >> 4) << 2) | ((kk & 7) >> 1);
  int h = kk & 1;
  base[(((size_t)(nt * KT + kt) * 32 + lane) << 4) + (j << 1) + h] = (bf16)v;
}

// A-matrix (M x K), 16-bit: lane row=lane&15; lanes>=16 carry K+8;
// VGPRs 0-3 -> K 0..7 (+offsets), VGPRs 4-7 -> K 16..23 (+offsets).
__device__ __forceinline__ void packA(bf16* base, int KT, int row, int k, float v) {
  int mt = row >> 4, r = row & 15;
  int kt = k >> 5, kk = k & 31;
  int lane = r + ((kk & 8) << 1);
  int j = ((kk >> 4) << 2) | ((kk & 7) >> 1);
  int h = kk & 1;
  base[(((size_t)(mt * KT + kt) * 32 + lane) << 4) + (j << 1) + h] = (bf16)v;
}

// One 16x16 output tile: loop K tiles of packed fragments (contiguous 32B/lane).
__device__ __forceinline__ v8f wmma_mm(const v16bf* A, const v16bf* Bm, int KT, int lane) {
  v8f acc = {};
  A += lane; Bm += lane;
  for (int kt = 0; kt < KT; ++kt) {
    acc = __builtin_amdgcn_wmma_f32_16x16x32_bf16(false, A[kt * 32], false, Bm[kt * 32],
                                                  (short)0, acc, false, false);
  }
  return acc;
}

// ---- one-time weight packing ----------------------------------------------
__global__ void precompute_kernel(const float* __restrict__ Wih1, const float* __restrict__ Whh1,
                                  const float* __restrict__ bih1, const float* __restrict__ bhh1,
                                  const float* __restrict__ Wih2, const float* __restrict__ Whh2,
                                  const float* __restrict__ bih2, const float* __restrict__ bhh2,
                                  const float* __restrict__ emb,
                                  bf16* __restrict__ W1p, bf16* __restrict__ W2p, bf16* __restrict__ EMBp,
                                  float* __restrict__ bias1, float* __restrict__ bias2) {
  const int NW1 = 1280 * 2048, NW2 = 768 * 1024, NE = 512 * 4096;
  const int total = NW1 + NW2 + NE + 2048 + 1024;
  for (int id = blockIdx.x * blockDim.x + threadIdx.x; id < total; id += gridDim.x * blockDim.x) {
    if (id < NW1) {                       // B1[k][n] = [W_ih1 | W_hh1][n][k]
      int k = id / 2048, n = id % 2048;
      float v = (k < 768) ? Wih1[(size_t)n * 768 + k] : Whh1[(size_t)n * 512 + (k - 768)];
      packB(W1p, KT1, k, n, v);
    } else if (id < NW1 + NW2) {          // B2[k][n] = [W_ih2 | W_hh2][n][k]
      int r = id - NW1;
      int k = r / 1024, n = r % 1024;
      float v = (k < 512) ? Wih2[(size_t)n * 512 + k] : Whh2[(size_t)n * 256 + (k - 512)];
      packB(W2p, KT2, k, n, v);
    } else if (id < NW1 + NW2 + NE) {     // Bout[k][n] = emb[n][k]
      int r = id - NW1 - NW2;
      int k = r / V_, n = r % V_;
      packB(EMBp, KTY, k, n, emb[(size_t)n * E_ + k]);
    } else {
      int r = id - NW1 - NW2 - NE;
      if (r < 2048) bias1[r] = bih1[r] + bhh1[r];
      else { int q = r - 2048; bias2[q] = bih2[q] + bhh2[q]; }
    }
  }
}

__global__ void zero_kernel(unsigned int* p, size_t nwords) {
  size_t i = (size_t)blockIdx.x * blockDim.x + threadIdx.x;
  size_t stride = (size_t)gridDim.x * blockDim.x;
  for (; i < nwords; i += stride) p[i] = 0u;
}

// ---- LSTM1: gates1 = x1 @ [Wih1|Whh1]^T + b ; fused activation ------------
// grid 32 blocks (16-col blocks of H1), 8 waves, 2 tiles/wave = 4 mt x 4 gates.
__global__ void lstm1_kernel(const bf16* __restrict__ xr, const bf16* __restrict__ w,
                             const float* __restrict__ bias, float* __restrict__ c1,
                             bf16* __restrict__ x1w, bf16* __restrict__ x2w) {
  __shared__ float g[4 * 64 * 16];
  int cb = blockIdx.x;
  int wv = threadIdx.x >> 5, lane = threadIdx.x & 31;
  const v16bf* X = (const v16bf*)xr;
  const v16bf* W = (const v16bf*)w;
  for (int i = 0; i < 2; ++i) {
    int p = wv * 2 + i;
    int mt = p & 3, gate = p >> 2;
    int nt = gate * 32 + cb;
    v8f acc = wmma_mm(X + (size_t)mt * KT1 * 32, W + (size_t)nt * KT1 * 32, KT1, lane);
    int col = lane & 15, ro = (lane >> 4) * 8;
#pragma unroll
    for (int r = 0; r < 8; ++r)
      g[(gate * 64 + mt * 16 + ro + r) * 16 + col] = acc[r];
  }
  __syncthreads();
#pragma unroll
  for (int i = 0; i < 4; ++i) {
    int o = threadIdx.x + i * 256;
    int b = o >> 4, col = o & 15;
    int n = cb * 16 + col;
    float gi = g[(0 * 64 + b) * 16 + col] + bias[0 * H1_ + n];
    float gf = g[(1 * 64 + b) * 16 + col] + bias[1 * H1_ + n];
    float gg = g[(2 * 64 + b) * 16 + col] + bias[2 * H1_ + n];
    float go = g[(3 * 64 + b) * 16 + col] + bias[3 * H1_ + n];
    float ii = sigmoidf_(gi), ff = sigmoidf_(gf), oo = sigmoidf_(go), tg = tanhf(gg);
    float cold = c1[b * H1_ + n];
    float cn = ff * cold + ii * tg;
    float h = oo * tanhf(cn);
    c1[b * H1_ + n] = cn;
    packA(x2w, KT2, b, n, h);              // h1 region of x2 (this step's LSTM2)
    packA(x1w, KT1, b, E_ + A_ + n, h);    // h1 region of next step's x1
  }
}

// ---- LSTM2: 16 blocks (16-col blocks of A=256) ----------------------------
__global__ void lstm2_kernel(const bf16* __restrict__ xr, const bf16* __restrict__ w,
                             const float* __restrict__ bias, float* __restrict__ c2,
                             float* __restrict__ h2, bf16* __restrict__ x2w,
                             bf16* __restrict__ ybuf) {
  __shared__ float g[4 * 64 * 16];
  int cb = blockIdx.x;
  int wv = threadIdx.x >> 5, lane = threadIdx.x & 31;
  const v16bf* X = (const v16bf*)xr;
  const v16bf* W = (const v16bf*)w;
  for (int i = 0; i < 2; ++i) {
    int p = wv * 2 + i;
    int mt = p & 3, gate = p >> 2;
    int nt = gate * 16 + cb;
    v8f acc = wmma_mm(X + (size_t)mt * KT2 * 32, W + (size_t)nt * KT2 * 32, KT2, lane);
    int col = lane & 15, ro = (lane >> 4) * 8;
#pragma unroll
    for (int r = 0; r < 8; ++r)
      g[(gate * 64 + mt * 16 + ro + r) * 16 + col] = acc[r];
  }
  __syncthreads();
#pragma unroll
  for (int i = 0; i < 4; ++i) {
    int o = threadIdx.x + i * 256;
    int b = o >> 4, col = o & 15;
    int n = cb * 16 + col;
    float gi = g[(0 * 64 + b) * 16 + col] + bias[0 * A_ + n];
    float gf = g[(1 * 64 + b) * 16 + col] + bias[1 * A_ + n];
    float gg = g[(2 * 64 + b) * 16 + col] + bias[2 * A_ + n];
    float go = g[(3 * 64 + b) * 16 + col] + bias[3 * A_ + n];
    float ii = sigmoidf_(gi), ff = sigmoidf_(gf), oo = sigmoidf_(go), tg = tanhf(gg);
    float cold = c2[b * A_ + n];
    float cn = ff * cold + ii * tg;
    float h = oo * tanhf(cn);
    c2[b * A_ + n] = cn;
    h2[b * A_ + n] = h;
    packA(x2w, KT2, b, H1_ + n, h);        // h2 region of next step's x2
    packA(ybuf, KTY, b, n, h);             // q region of y (this step's output GEMM)
  }
}

// ---- attention: energy / masked softmax / context (exact f32) -------------
__global__ void attn_kernel(const float* __restrict__ kk, const float* __restrict__ vv,
                            const float* __restrict__ h2, const int* __restrict__ elen,
                            bf16* __restrict__ x1w, bf16* __restrict__ ybuf) {
  __shared__ __align__(16) float q[A_];
  __shared__ float e[TD_];
  __shared__ float red[256];
  int b = blockIdx.x, tid = threadIdx.x;
  q[tid] = h2[b * A_ + tid];
  __syncthreads();
  int len = elen[b];
  const float* kb = kk + (size_t)b * TD_ * A_;
  for (int half = 0; half < 2; ++half) {
    int t = tid + half * 256;
    const float4* kr = (const float4*)(kb + (size_t)t * A_);
    const float4* qr = (const float4*)q;
    float s = 0.f;
#pragma unroll 8
    for (int i = 0; i < A_ / 4; ++i) {
      float4 kv = kr[i]; float4 qv = qr[i];
      s += kv.x * qv.x + kv.y * qv.y + kv.z * qv.z + kv.w * qv.w;
    }
    if (t >= len) s = -1000000000.0f;
    e[t] = s;
  }
  __syncthreads();
  red[tid] = fmaxf(e[tid], e[tid + 256]);
  __syncthreads();
  for (int s = 128; s > 0; s >>= 1) {
    if (tid < s) red[tid] = fmaxf(red[tid], red[tid + s]);
    __syncthreads();
  }
  float mx = red[0];
  __syncthreads();
  float e0 = expf(e[tid] - mx);
  float e1 = expf(e[tid + 256] - mx);
  e[tid] = e0; e[tid + 256] = e1;
  red[tid] = e0 + e1;
  __syncthreads();
  for (int s = 128; s > 0; s >>= 1) {
    if (tid < s) red[tid] += red[tid + s];
    __syncthreads();
  }
  float inv = 1.f / red[0];
  const float* vb = vv + (size_t)b * TD_ * A_;
  float ctx = 0.f;
#pragma unroll 4
  for (int t = 0; t < TD_; ++t) ctx += e[t] * vb[(size_t)t * A_ + tid];
  ctx *= inv;
  packA(x1w, KT1, b, E_ + tid, ctx);       // ctx region of next step's x1
  packA(ybuf, KTY, b, A_ + tid, ctx);      // ctx region of y (this step)
}

// ---- output GEMM: logits = y @ emb^T + bias; 64 blocks x 8 waves x 2 tiles
__global__ void outgemm_kernel(const bf16* __restrict__ y, const bf16* __restrict__ w,
                               const float* __restrict__ obias, float* __restrict__ logits,
                               float* __restrict__ dout, int tstep) {
  int wv = threadIdx.x >> 5, lane = threadIdx.x & 31;
  const v16bf* Y = (const v16bf*)y;
  const v16bf* W = (const v16bf*)w;
  for (int i = 0; i < 2; ++i) {
    int idx = (blockIdx.x * 8 + wv) * 2 + i;
    int mt = idx & 3, nt = idx >> 2;
    v8f acc = wmma_mm(Y + (size_t)mt * KTY * 32, W + (size_t)nt * KTY * 32, KTY, lane);
    int col = lane & 15, ro = (lane >> 4) * 8;
    int n = nt * 16 + col;
    float bb = obias[n];
#pragma unroll
    for (int r = 0; r < 8; ++r) {
      int b = mt * 16 + ro + r;
      float val = acc[r] + bb;
      logits[(size_t)b * V_ + n] = val;
      dout[((size_t)b * V_ + n) * TO_ + tstep] = val;   // (B, V, TO)
    }
  }
}

// ---- argmax (first-max, matching jnp.argmax) + embedding pack -------------
__global__ void argmax_embed_kernel(const float* __restrict__ logits, const float* __restrict__ emb,
                                    int* __restrict__ chars, bf16* __restrict__ x1w) {
  __shared__ float bv[256];
  __shared__ int   bi[256];
  __shared__ int   ch;
  int b = blockIdx.x, tid = threadIdx.x;
  const float* row = logits + (size_t)b * V_;
  float best = -INFINITY; int bid = 0;
#pragma unroll 4
  for (int i = 0; i < 16; ++i) {
    float v = row[tid * 16 + i];
    if (v > best) { best = v; bid = tid * 16 + i; }     // first occurrence in chunk
  }
  bv[tid] = best; bi[tid] = bid;
  __syncthreads();
  for (int s = 128; s > 0; s >>= 1) {
    if (tid < s && bv[tid + s] > bv[tid]) { bv[tid] = bv[tid + s]; bi[tid] = bi[tid + s]; }
    __syncthreads();                                    // ties keep lower index
  }
  if (tid == 0) { ch = bi[0]; chars[b] = ch; }
  __syncthreads();
  const float* er = emb + (size_t)ch * E_;
  packA(x1w, KT1, b, tid, er[tid]);                     // ex region of next x1
  packA(x1w, KT1, b, tid + 256, er[tid + 256]);
}

extern "C" void kernel_launch(void* const* d_in, const int* in_sizes, int n_in,
                              void* d_out, int out_size, void* d_ws, size_t ws_size,
                              hipStream_t stream) {
  const float* kin   = (const float*)d_in[0];
  const float* vin   = (const float*)d_in[1];
  const float* emb   = (const float*)d_in[2];
  const float* Wih1  = (const float*)d_in[3];
  const float* Whh1  = (const float*)d_in[4];
  const float* bih1  = (const float*)d_in[5];
  const float* bhh1  = (const float*)d_in[6];
  const float* Wih2  = (const float*)d_in[7];
  const float* Whh2  = (const float*)d_in[8];
  const float* bih2  = (const float*)d_in[9];
  const float* bhh2  = (const float*)d_in[10];
  const float* obias = (const float*)d_in[11];
  const int*   elen  = (const int*)d_in[12];
  // d_in[13] = gt, d_in[14] = p_tf: unused (p_tf == 0 -> pure argmax feedback)
  float* dout = (float*)d_out;

  char* ws = (char*)d_ws;
  size_t off = 0;
  auto alloc = [&](size_t sz) { void* p = ws + off; off += (sz + 255) & ~(size_t)255; return p; };

  // static (packed once)
  bf16*  W1p   = (bf16*)alloc((size_t)1280 * 2048 * 2);
  bf16*  W2p   = (bf16*)alloc((size_t)768 * 1024 * 2);
  bf16*  EMBp  = (bf16*)alloc((size_t)512 * 4096 * 2);
  float* bias1 = (float*)alloc(2048 * 4);
  float* bias2 = (float*)alloc(1024 * 4);
  // dynamic (zeroed each launch)
  char* zstart = ws + off;
  bf16*  X1A   = (bf16*)alloc((size_t)B_ * 1280 * 2);   // x1 double buffer
  bf16*  X1B   = (bf16*)alloc((size_t)B_ * 1280 * 2);
  bf16*  X2A   = (bf16*)alloc((size_t)B_ * 768 * 2);    // x2 double buffer
  bf16*  X2B   = (bf16*)alloc((size_t)B_ * 768 * 2);
  bf16*  Y     = (bf16*)alloc((size_t)B_ * 512 * 2);
  float* c1    = (float*)alloc((size_t)B_ * H1_ * 4);
  float* h2f   = (float*)alloc((size_t)B_ * A_ * 4);
  float* c2    = (float*)alloc((size_t)B_ * A_ * 4);
  float* logits= (float*)alloc((size_t)B_ * V_ * 4);
  int*   chars = (int*)alloc(B_ * 4);
  size_t zwords = ((size_t)(ws + off - zstart)) / 4;

  zero_kernel<<<512, 256, 0, stream>>>((unsigned int*)zstart, zwords);
  precompute_kernel<<<2048, 256, 0, stream>>>(Wih1, Whh1, bih1, bhh1,
                                              Wih2, Whh2, bih2, bhh2,
                                              emb, W1p, W2p, EMBp, bias1, bias2);

  for (int t = 0; t < TO_; ++t) {
    bf16* x1r = (t & 1) ? X1B : X1A;
    bf16* x1w = (t & 1) ? X1A : X1B;
    bf16* x2r = (t & 1) ? X2B : X2A;
    bf16* x2w = (t & 1) ? X2A : X2B;
    lstm1_kernel<<<32, 256, 0, stream>>>(x1r, W1p, bias1, c1, x1w, x2r);
    lstm2_kernel<<<16, 256, 0, stream>>>(x2r, W2p, bias2, c2, h2f, x2w, Y);
    attn_kernel<<<64, 256, 0, stream>>>(kin, vin, h2f, elen, x1w, Y);
    outgemm_kernel<<<64, 256, 0, stream>>>(Y, EMBp, obias, logits, dout, t);
    argmax_embed_kernel<<<64, 256, 0, stream>>>(logits, emb, chars, x1w);
  }
}
